// MultiHeadedAttention_73710228734780
// MI455X (gfx1250) — matrix operand order repaired and checked by reading, not compiled, not run on hardware
//
#include <hip/hip_runtime.h>
#include <hip/hip_bf16.h>

// MHA: B=2, T=2048, D=1024, H=16, DK=64.  All matmuls via v_wmma_f32_16x16x32_bf16.
// GEMMs: register-blocked 64x32 per wave (8 WMMA : 6 fragment loads per K-step).
// Attention: 4 q-tile waves per block share K / V^T chunks through LDS.

#define BB 2
#define TT 2048
#define DD 1024
#define HH 16
#define DKK 64

typedef __attribute__((ext_vector_type(16))) __bf16 v16bf;
typedef __attribute__((ext_vector_type(8)))  __bf16 v8bf;
typedef __attribute__((ext_vector_type(8)))  float  v8f;

union V16U { v16bf v; v8bf h[2]; };

__device__ __forceinline__ v8bf ld8(const __bf16* p) { return *(const v8bf*)p; }

__device__ __forceinline__ v16bf mk16(v8bf lo, v8bf hi) {
    V16U u; u.h[0] = lo; u.h[1] = hi; return u.v;
}

__device__ __forceinline__ v8f wmma_bf16(v16bf a, v16bf b, v8f c) {
    return __builtin_amdgcn_wmma_f32_16x16x32_bf16(false, a, false, b, (short)0, c, false, false);
}

// ---------------------------------------------------------------- f32 -> bf16
__global__ void mha_cvt_bf16(const float* __restrict__ in, __bf16* __restrict__ out, int n) {
    int i = blockIdx.x * blockDim.x + threadIdx.x;
    if (i < n) out[i] = (__bf16)in[i];
}

// ------------------------------------------- register-blocked WMMA GEMM + bias
// Y[m][n] = sum_k X[m][k] * W[n][k] + bias[n]   (torch Linear: x @ W.T + b)
// Each wave: 64x32 output (4 M-tiles x 2 N-tiles), 8 WMMAs per 32-K step.
// mode 0: bf16 head-major [B,H,T,DK]; mode 1: f32 row-major [M,N];
// mode 2: bf16 transposed [B,H,DK,T]  (for V, so attention reads V^T contiguously)
#define GM 4
#define GN 2
#define GWAVES 4
__global__ __launch_bounds__(32 * GWAVES)
void mha_gemm(const __bf16* __restrict__ X, const __bf16* __restrict__ W,
              const float* __restrict__ bias, __bf16* __restrict__ outb,
              float* __restrict__ outf, int Kc, int N, int mode) {
    const int lane = threadIdx.x & 31;
    const int wave = threadIdx.x >> 5;
    const int m0 = blockIdx.x * (16 * GM);
    const int n0 = (blockIdx.y * GWAVES + wave) * (16 * GN);
    const int hh = lane >> 4;
    const int c  = lane & 15;

    const __bf16* xr[GM];
    const __bf16* wr[GN];
#pragma unroll
    for (int mt = 0; mt < GM; ++mt) xr[mt] = X + (size_t)(m0 + mt * 16 + c) * Kc;
#pragma unroll
    for (int nt = 0; nt < GN; ++nt) wr[nt] = W + (size_t)(n0 + nt * 16 + c) * Kc;

    v8f acc[GM][GN] = {};
    for (int k0 = 0; k0 < Kc; k0 += 32) {
        v16bf a[GM], b[GN];
#pragma unroll
        for (int mt = 0; mt < GM; ++mt)
            a[mt] = mk16(ld8(xr[mt] + k0 + hh * 8), ld8(xr[mt] + k0 + 16 + hh * 8));
#pragma unroll
        for (int nt = 0; nt < GN; ++nt)
            b[nt] = mk16(ld8(wr[nt] + k0 + hh * 8), ld8(wr[nt] + k0 + 16 + hh * 8));
#pragma unroll
        for (int mt = 0; mt < GM; ++mt)
#pragma unroll
            for (int nt = 0; nt < GN; ++nt)
                acc[mt][nt] = wmma_bf16(a[mt], b[nt], acc[mt][nt]);
    }

#pragma unroll
    for (int nt = 0; nt < GN; ++nt) {
        const int col = n0 + nt * 16 + c;
        const float bv = bias[col];
        const int head = col >> 6, dk = col & 63;
#pragma unroll
        for (int mt = 0; mt < GM; ++mt) {
#pragma unroll
            for (int r = 0; r < 8; ++r) {
                const int row = m0 + mt * 16 + hh * 8 + r;      // row = b*T + t
                const float val = acc[mt][nt][r] + bv;
                if (mode == 0) {
                    const int bb = row >> 11, t = row & (TT - 1);
                    outb[(((size_t)bb * HH + head) * TT + t) * DKK + dk] = (__bf16)val;
                } else if (mode == 1) {
                    outf[(size_t)row * N + col] = val;
                } else {
                    const int bb = row >> 11, t = row & (TT - 1);
                    outb[(((size_t)bb * HH + head) * DKK + dk) * TT + t] = (__bf16)val;
                }
            }
        }
    }
}

// ------------------------------------------------------- flash attention core
// Block = 4 waves = 4 query tiles of one (b,head); K / V^T chunks staged in LDS.
// Sᵀ = K_tile * Qᵀ so the score D-layout coincides with the P A-operand layout.
__global__ __launch_bounds__(128)
void mha_attn(const __bf16* __restrict__ Qp, const __bf16* __restrict__ Kp,
              const __bf16* __restrict__ Vt, const int* __restrict__ mask,
              __bf16* __restrict__ attnout) {
    __shared__ __bf16 Ks[32 * 64];   // [key][dk], 4KB
    __shared__ __bf16 Vs[64 * 32];   // [dk][key], 4KB

    const int bh = blockIdx.x;                 // b*H + head
    const int b = bh >> 4, head = bh & 15;
    const int tid = threadIdx.x;
    const int lane = tid & 31, wave = tid >> 5;
    const int qt = blockIdx.y * 4 + wave;
    const int hh = lane >> 4, c = lane & 15;

    const __bf16* Kb  = Kp + (size_t)bh * TT * DKK;   // [key][dk]
    const __bf16* Vtb = Vt + (size_t)bh * DKK * TT;   // [dk][key]
    const int* mrow = mask + b * TT;

    // Q as B-operand of Sᵀ: lane = query col c, elems = dk hh*8+0..7, 16+hh*8+0..7
    const __bf16* qr = Qp + ((size_t)bh * TT + qt * 16 + c) * DKK;
    const v16bf qb0 = mk16(ld8(qr + hh * 8), ld8(qr + 16 + hh * 8));           // dk 0..31
    const v16bf qb1 = mk16(ld8(qr + 32 + hh * 8), ld8(qr + 48 + hh * 8)); // dk 32..63

    v8f o[4] = {{}, {}, {}, {}};               // O tile 16q x 64dk, f32
    float m_run = -3.0e38f, l_run = 0.0f;
    const float scale = 0.125f;                // 1/sqrt(64)

    for (int kc = 0; kc < TT; kc += 32) {
        // cooperative stage: 128 threads x 32B each for K and V^T chunks
        {
            const __bf16* ks = Kb + (size_t)(kc + (tid >> 2)) * DKK + (tid & 3) * 16;
            *(v8bf*)(Ks + tid * 16)     = ld8(ks);
            *(v8bf*)(Ks + tid * 16 + 8) = ld8(ks + 8);
            const __bf16* vs = Vtb + (size_t)(tid >> 1) * TT + kc + (tid & 1) * 16;
            *(v8bf*)(Vs + tid * 16)     = ld8(vs);
            *(v8bf*)(Vs + tid * 16 + 8) = ld8(vs + 8);
            if (kc + 32 < TT) {
                __builtin_prefetch(ks + 32 * DKK, 0, 1);
                __builtin_prefetch(vs + 32, 0, 1);
            }
        }
        __syncthreads();

        // K rows as A-operand from LDS (M = key index)
        const __bf16* kr0 = Ks + c * 64;
        const __bf16* kr1 = Ks + (16 + c) * 64;
        v16bf ka0 = mk16(ld8(kr0 + hh * 8), ld8(kr0 + 16 + hh * 8));
        v16bf ka1 = mk16(ld8(kr0 + 32 + hh * 8), ld8(kr0 + 48 + hh * 8));
        v16bf kb0 = mk16(ld8(kr1 + hh * 8), ld8(kr1 + 16 + hh * 8));
        v16bf kb1 = mk16(ld8(kr1 + 32 + hh * 8), ld8(kr1 + 48 + hh * 8));

        v8f z = {};
        v8f s0 = wmma_bf16(ka0, qb0, z); s0 = wmma_bf16(ka1, qb1, s0); // keys kc..kc+15
        v8f s1 = wmma_bf16(kb0, qb0, z); s1 = wmma_bf16(kb1, qb1, s1); // keys kc+16..kc+31

        // scale + mask; lane holds query col c, VGPR r = key hh*8+r (+16 for s1)
        float cmax = -3.0e38f;
#pragma unroll
        for (int r = 0; r < 8; ++r) {
            float v0 = mrow[kc + hh * 8 + r]      ? s0[r] * scale : -3.0e38f;
            float v1 = mrow[kc + 16 + hh * 8 + r] ? s1[r] * scale : -3.0e38f;
            s0[r] = v0; s1[r] = v1;
            cmax = fmaxf(cmax, fmaxf(v0, v1));
        }
        cmax = fmaxf(cmax, __shfl_xor(cmax, 16, 32));   // merge lane halves (same query)
        const float m_new = fmaxf(m_run, cmax);
        const float alpha = __expf(m_run - m_new);

        // P = exp(s - m); score D-layout == P A-operand layout -> in-lane convert
        v16bf pa;
        float csum = 0.0f;
#pragma unroll
        for (int r = 0; r < 8; ++r) {
            float p0 = __expf(s0[r] - m_new);
            float p1 = __expf(s1[r] - m_new);
            csum += p0 + p1;
            pa[r]     = (__bf16)p0;
            pa[8 + r] = (__bf16)p1;
        }
        csum += __shfl_xor(csum, 16, 32);
        l_run = l_run * alpha + csum;
        m_run = m_new;

        // redistribute alpha from query-column layout to O's query-row layout
        float af[8];
#pragma unroll
        for (int r = 0; r < 8; ++r) af[r] = __shfl(alpha, hh * 8 + r, 32);
#pragma unroll
        for (int t = 0; t < 4; ++t)
#pragma unroll
            for (int r = 0; r < 8; ++r) o[t][r] *= af[r];

        // V^T as B-operand from LDS: lane = dk col t*16+c, elems = keys hh*8+0..7, 16+hh*8+0..7
#pragma unroll
        for (int t = 0; t < 4; ++t) {
            const __bf16* vr = Vs + (t * 16 + c) * 32 + hh * 8;
            v16bf vb = mk16(ld8(vr), ld8(vr + 16));
            o[t] = wmma_bf16(pa, vb, o[t]);
        }
        __syncthreads();
    }

    // normalize rows by l and store bf16 to [B,T,D] row-major
    float lf[8];
#pragma unroll
    for (int r = 0; r < 8; ++r) lf[r] = 1.0f / fmaxf(__shfl(l_run, hh * 8 + r, 32), 1e-30f);
#pragma unroll
    for (int t = 0; t < 4; ++t)
#pragma unroll
        for (int r = 0; r < 8; ++r) {
            const int row = b * TT + qt * 16 + hh * 8 + r;
            attnout[(size_t)row * DD + head * 64 + t * 16 + c] = (__bf16)(o[t][r] * lf[r]);
        }
}

// ---------------------------------------------------------------------- launch
extern "C" void kernel_launch(void* const* d_in, const int* in_sizes, int n_in,
                              void* d_out, int out_size, void* d_ws, size_t ws_size,
                              hipStream_t stream) {
    const float* q  = (const float*)d_in[0];
    const float* k  = (const float*)d_in[1];
    const float* v  = (const float*)d_in[2];
    const int*   mk = (const int*)d_in[3];
    const float* Wq = (const float*)d_in[4];  const float* bq = (const float*)d_in[5];
    const float* Wk = (const float*)d_in[6];  const float* bk = (const float*)d_in[7];
    const float* Wv = (const float*)d_in[8];  const float* bv = (const float*)d_in[9];
    const float* Wo = (const float*)d_in[10]; const float* bo = (const float*)d_in[11];
    float* out = (float*)d_out;

    const size_t XSZ = (size_t)BB * TT * DD;   // 4M elems
    const size_t WSZ = (size_t)DD * DD;        // 1M elems
    __bf16* Xq  = (__bf16*)d_ws;               // needs 7*XSZ + 4*WSZ bf16 = 64MB
    __bf16* Xk  = Xq  + XSZ;
    __bf16* Xv  = Xk  + XSZ;
    __bf16* Wqb = Xv  + XSZ;
    __bf16* Wkb = Wqb + WSZ;
    __bf16* Wvb = Wkb + WSZ;
    __bf16* Wob = Wvb + WSZ;
    __bf16* Qp  = Wob + WSZ;                   // [B,H,T,DK]
    __bf16* Kp  = Qp  + XSZ;                   // [B,H,T,DK]
    __bf16* Vtp = Kp  + XSZ;                   // [B,H,DK,T]  (transposed)
    __bf16* AO  = Vtp + XSZ;                   // [B,T,D]

    const int cb = 256;
    mha_cvt_bf16<<<(int)((XSZ + cb - 1) / cb), cb, 0, stream>>>(q,  Xq,  (int)XSZ);
    mha_cvt_bf16<<<(int)((XSZ + cb - 1) / cb), cb, 0, stream>>>(k,  Xk,  (int)XSZ);
    mha_cvt_bf16<<<(int)((XSZ + cb - 1) / cb), cb, 0, stream>>>(v,  Xv,  (int)XSZ);
    mha_cvt_bf16<<<(int)((WSZ + cb - 1) / cb), cb, 0, stream>>>(Wq, Wqb, (int)WSZ);
    mha_cvt_bf16<<<(int)((WSZ + cb - 1) / cb), cb, 0, stream>>>(Wk, Wkb, (int)WSZ);
    mha_cvt_bf16<<<(int)((WSZ + cb - 1) / cb), cb, 0, stream>>>(Wv, Wvb, (int)WSZ);
    mha_cvt_bf16<<<(int)((WSZ + cb - 1) / cb), cb, 0, stream>>>(Wo, Wob, (int)WSZ);

    // GEMMs: M = 4096 -> 64 blocks of 64 rows; N = 1024 -> 8 blocks of 4 waves x 32 cols
    dim3 ggrid(64, 8), gblk(32 * GWAVES);
    mha_gemm<<<ggrid, gblk, 0, stream>>>(Xq, Wqb, bq, Qp,  nullptr, DD, DD, 0);
    mha_gemm<<<ggrid, gblk, 0, stream>>>(Xk, Wkb, bk, Kp,  nullptr, DD, DD, 0);
    mha_gemm<<<ggrid, gblk, 0, stream>>>(Xv, Wvb, bv, Vtp, nullptr, DD, DD, 2);

    // attention: 4 waves (4 q-tiles) per block, LDS-shared K/V^T chunks
    dim3 agrid(BB * HH, TT / 16 / 4);
    mha_attn<<<agrid, 128, 0, stream>>>(Qp, Kp, Vtp, mk, AO);

    // output projection -> f32 d_out
    mha_gemm<<<ggrid, gblk, 0, stream>>>(AO, Wob, bo, nullptr, out, DD, DD, 1);
}
